// AttentionRetrieval_53094385713849
// MI455X (gfx1250) — compile-verified, matrix operand order checked by reading, not run on hardware
//
#include <hip/hip_runtime.h>
#include <hip/hip_bf16.h>

typedef __bf16 bf16_t;
typedef __attribute__((ext_vector_type(8)))  __bf16 v8bf;
typedef __attribute__((ext_vector_type(16))) __bf16 v16bf;
typedef __attribute__((ext_vector_type(8)))  float  v8f;

#define HDIM 128
#define DH   4096   /* 32 * 128 flattened feature length */
#define NQ   4096
#define NS   16384
#define TOPK 16

#define KSTAGE  128            /* K elements staged in LDS per buffer (4 WMMA k-blocks) */
#define NSTAGES (DH / KSTAGE)  /* 32 */

__device__ __forceinline__ v16bf cat8(v8bf lo, v8bf hi) {
  return __builtin_shufflevector(lo, hi, 0,1,2,3,4,5,6,7,8,9,10,11,12,13,14,15);
}

// ---------------------------------------------------------------------------
// Projection: Y[r,o] = sum_h X[r,h]*W[o,h] + b[o], output rounded to bf16.
// One block = 16 rows x 128 cols; 8 waves, each wave owns one 16x16 N-tile.
// ---------------------------------------------------------------------------
__global__ void __launch_bounds__(256) proj_bf16_kernel(
    const float* __restrict__ X, const float* __restrict__ W,
    const float* __restrict__ b, bf16_t* __restrict__ Y) {
  __shared__ bf16_t Ws[HDIM * HDIM];   // 32 KB
  __shared__ bf16_t Xs[16 * HDIM];     //  4 KB
  const int tid = threadIdx.x;
  const size_t rowbase = (size_t)blockIdx.x * 16;
  for (int i = tid; i < HDIM * HDIM; i += 256) Ws[i] = (bf16_t)W[i];
  for (int i = tid; i < 16 * HDIM;   i += 256) Xs[i] = (bf16_t)X[rowbase * HDIM + i];
  __syncthreads();

  const int lane = tid & 31, wv = tid >> 5;
  const int col  = lane & 15;          // A row index AND B/D column index
  const int half = lane >> 4;
  const int off  = half * 8;           // K sub-offset per ISA bf16 layout
  const bf16_t* arow = &Xs[col * HDIM];
  const bf16_t* wrow = &Ws[(wv * 16 + col) * HDIM];   // B col n = W row n

  v8f c = {};
#pragma unroll
  for (int kb = 0; kb < 4; ++kb) {
    const int k0 = kb * 32 + off;
    v16bf a  = cat8(*(const v8bf*)&arow[k0], *(const v8bf*)&arow[k0 + 16]);
    v16bf bm = cat8(*(const v8bf*)&wrow[k0], *(const v8bf*)&wrow[k0 + 16]);
    c = __builtin_amdgcn_wmma_f32_16x16x32_bf16(false, a, false, bm,
                                                (short)0, c, false, false);
  }
  const float bias = b[wv * 16 + col];
#pragma unroll
  for (int v = 0; v < 8; ++v) {
    const int m = v + half * 8;        // D tile row per ISA layout
    Y[(rowbase + m) * HDIM + wv * 16 + col] = (bf16_t)(c[v] + bias);
  }
}

// ---------------------------------------------------------------------------
// Row norms over the bf16-rounded projections (keeps the distance identity
// consistent with the bf16 WMMA cross term).
// ---------------------------------------------------------------------------
__global__ void __launch_bounds__(256) rownorm_kernel(
    const bf16_t* __restrict__ Y, float* __restrict__ out) {
  __shared__ float red[256];
  const int tid = threadIdx.x;
  const size_t row = blockIdx.x;
  float s = 0.f;
  for (int i = tid; i < DH; i += 256) {
    float v = (float)Y[row * DH + i];
    s += v * v;
  }
  red[tid] = s; __syncthreads();
  for (int st = 128; st > 0; st >>= 1) {
    if (tid < st) red[tid] += red[tid + st];
    __syncthreads();
  }
  if (tid == 0) out[row] = red[0];
}

// ---------------------------------------------------------------------------
// Main GEMM + distance epilogue:
//   sim[m,n] = (2*q[m]·k[n] - |q[m]|^2 - |k[n]|^2) / sqrt(128)
// Block tile 128(M) x 64(N); 8 waves stacked in M, each wave = 16x64 via
// 4 accumulators. The shared B panel (64 kproj rows) is double-buffered in
// LDS via CDNA5 async copies (global_load_async_to_lds_b128 / ASYNCcnt),
// overlapping the next stage's copy with the current stage's 16 WMMAs and
// eliminating the 8x duplicated global B traffic.
// ---------------------------------------------------------------------------
__global__ void __launch_bounds__(256) sim_kernel(
    const bf16_t* __restrict__ Q, const bf16_t* __restrict__ Kp,
    const float* __restrict__ q2, const float* __restrict__ k2,
    float* __restrict__ sim) {
  __shared__ bf16_t Bs[2][64 * KSTAGE];   // 2 x 16 KB, [col][kk] per buffer
  const int tid = threadIdx.x, lane = tid & 31, wv = tid >> 5;
  const int col  = lane & 15;
  const int half = lane >> 4;
  const int off  = half * 8;
  const int mbase = blockIdx.x * 128 + wv * 16;
  const int nbase = blockIdx.y * 64;

  const bf16_t* qrow = Q + (size_t)(mbase + col) * DH;   // A row for this lane

  // Issue the async fill of one 64x128 B panel: 16 KB, 4 x b128 per thread.
  auto issue_stage = [&](int s, int buf) {
#pragma unroll
    for (int j = 0; j < 4; ++j) {
      const int cid = tid * 4 + j;       // 16-byte chunk id, 0..1023
      const int c   = cid >> 4;          // column 0..63 (16 chunks per column)
      const int kk  = (cid & 15) * 8;    // element offset within column
      const unsigned lds_off =
          (unsigned)(unsigned long long)(&Bs[buf][c * KSTAGE + kk]);
      const unsigned long long gaddr =
          (unsigned long long)(const void*)(Kp + (size_t)(nbase + c) * DH +
                                            (size_t)s * KSTAGE + kk);
      asm volatile("global_load_async_to_lds_b128 %0, %1, off"
                   :: "v"(lds_off), "v"(gaddr) : "memory");
    }
  };

  issue_stage(0, 0);

  v8f c0 = {}, c1 = {}, c2 = {}, c3 = {};
  for (int s = 0; s < NSTAGES; ++s) {
    const int buf = s & 1;
    if (s + 1 < NSTAGES) {
      issue_stage(s + 1, buf ^ 1);                      // prefetch next panel
      __builtin_prefetch(qrow + (size_t)(s + 1) * KSTAGE, 0, 0);
      asm volatile("s_wait_asynccnt 0x4" ::: "memory"); // this panel landed
    } else {
      asm volatile("s_wait_asynccnt 0x0" ::: "memory");
    }
    __syncthreads();                                    // panel visible to all

    const bf16_t* bbase = &Bs[buf][0];
#pragma unroll
    for (int kbl = 0; kbl < 4; ++kbl) {
      const int k0 = kbl * 32 + off;
      const size_t gk = (size_t)s * KSTAGE + k0;
      v16bf a = cat8(*(const v8bf*)&qrow[gk], *(const v8bf*)&qrow[gk + 16]);
      const bf16_t* p0 = bbase + (col +  0) * KSTAGE + k0;
      const bf16_t* p1 = bbase + (col + 16) * KSTAGE + k0;
      const bf16_t* p2 = bbase + (col + 32) * KSTAGE + k0;
      const bf16_t* p3 = bbase + (col + 48) * KSTAGE + k0;
      v16bf b0 = cat8(*(const v8bf*)&p0[0], *(const v8bf*)&p0[16]);
      c0 = __builtin_amdgcn_wmma_f32_16x16x32_bf16(false, a, false, b0, (short)0, c0, false, false);
      v16bf b1 = cat8(*(const v8bf*)&p1[0], *(const v8bf*)&p1[16]);
      c1 = __builtin_amdgcn_wmma_f32_16x16x32_bf16(false, a, false, b1, (short)0, c1, false, false);
      v16bf b2 = cat8(*(const v8bf*)&p2[0], *(const v8bf*)&p2[16]);
      c2 = __builtin_amdgcn_wmma_f32_16x16x32_bf16(false, a, false, b2, (short)0, c2, false, false);
      v16bf b3 = cat8(*(const v8bf*)&p3[0], *(const v8bf*)&p3[16]);
      c3 = __builtin_amdgcn_wmma_f32_16x16x32_bf16(false, a, false, b3, (short)0, c3, false, false);
    }
    __syncthreads();   // all reads done before this buffer is refilled (s+2)
  }

  const float scale = 0.08838834764831845f;  // 1/sqrt(128)
  const float kn0 = k2[nbase +  0 + col];
  const float kn1 = k2[nbase + 16 + col];
  const float kn2 = k2[nbase + 32 + col];
  const float kn3 = k2[nbase + 48 + col];
#pragma unroll
  for (int v = 0; v < 8; ++v) {
    const int gm = mbase + v + half * 8;
    const float qn = q2[gm];
    float* srow = sim + (size_t)gm * NS + nbase;
    srow[ 0 + col] = (2.f * c0[v] - qn - kn0) * scale;
    srow[16 + col] = (2.f * c1[v] - qn - kn1) * scale;
    srow[32 + col] = (2.f * c2[v] - qn - kn2) * scale;
    srow[48 + col] = (2.f * c3[v] - qn - kn3) * scale;
  }
}

// ---------------------------------------------------------------------------
// Top-16 + softmax per query row. Row (64KB) lives in dynamic LDS (320KB/WGP
// budget on CDNA5). 16 iterative argmax passes -> naturally sorted desc with
// lowest-index tie-break (deterministic, matches torch.topk ordering).
// ---------------------------------------------------------------------------
__global__ void __launch_bounds__(256) topk_kernel(
    const float* __restrict__ sim, int* __restrict__ idx_out,
    float* __restrict__ w_out) {
  extern __shared__ float vals[];                // NS floats = 64 KB
  __shared__ float rv[256];
  __shared__ int   ri[256];
  __shared__ float topv[TOPK];
  __shared__ int   topi[TOPK];
  const int tid = threadIdx.x;
  const size_t row = blockIdx.x;
  const float* srow = sim + row * NS;
  for (int i = tid; i < NS; i += 256) vals[i] = srow[i];
  __syncthreads();

  for (int sel = 0; sel < TOPK; ++sel) {
    float bv = -3.4e38f; int bi = NS;
    for (int i = tid; i < NS; i += 256) {        // ascending scan: '>' keeps
      float v = vals[i];                         // lowest index among equals
      if (v > bv) { bv = v; bi = i; }
    }
    rv[tid] = bv; ri[tid] = bi;
    __syncthreads();
    for (int st = 128; st > 0; st >>= 1) {
      if (tid < st) {
        const float ov = rv[tid + st]; const int oi = ri[tid + st];
        if (ov > rv[tid] || (ov == rv[tid] && oi < ri[tid])) { rv[tid] = ov; ri[tid] = oi; }
      }
      __syncthreads();
    }
    if (tid == 0) { topv[sel] = rv[0]; topi[sel] = ri[0]; vals[ri[0]] = -3.4e38f; }
    __syncthreads();
  }

  if (tid == 0) {
    float e[TOPK], s = 0.f;
    const float mx = topv[0];                    // sorted desc -> max first
    for (int i = 0; i < TOPK; ++i) { e[i] = __expf(topv[i] - mx); s += e[i]; }
    const float inv = 1.f / s;
    for (int i = 0; i < TOPK; ++i) {
      idx_out[row * TOPK + i] = topi[i];
      w_out[row * TOPK + i]   = e[i] * inv;
    }
  }
}

// ---------------------------------------------------------------------------
extern "C" void kernel_launch(void* const* d_in, const int* in_sizes, int n_in,
                              void* d_out, int out_size, void* d_ws, size_t ws_size,
                              hipStream_t stream) {
  (void)in_sizes; (void)n_in; (void)out_size; (void)ws_size;
  const float* query   = (const float*)d_in[0];  // (4096, 32, 128)
  const float* support = (const float*)d_in[1];  // (16384, 32, 128)
  const float* Wq      = (const float*)d_in[2];  // (128, 128)
  const float* bq      = (const float*)d_in[3];  // (128,)
  const float* Wk      = (const float*)d_in[4];  // (128, 128)
  const float* bk      = (const float*)d_in[5];  // (128,)
  // d_in[6] = k (== 16, baked into TOPK)

  // Workspace layout (bytes):
  //   qb  : NQ*DH*2          =  32 MB   @ 0
  //   kb  : NS*DH*2          = 128 MB   @ 32 MB
  //   sim : NQ*NS*4          = 256 MB   @ 160 MB
  //   q2  : NQ*4             =  16 KB   @ 416 MB
  //   k2  : NS*4             =  64 KB   @ 417 MB
  char* ws = (char*)d_ws;
  bf16_t* qb  = (bf16_t*)(ws);
  bf16_t* kb  = (bf16_t*)(ws + ((size_t)32  << 20));
  float*  sim = (float*) (ws + ((size_t)160 << 20));
  float*  q2  = (float*) (ws + ((size_t)416 << 20));
  float*  k2  = (float*) (ws + ((size_t)417 << 20));

  int*   idx_out = (int*)d_out;                       // first output (int32)
  float* w_out   = (float*)d_out + (size_t)NQ * TOPK; // second output (f32)

  // 1) Projections (bf16 outputs), 16 rows per block.
  proj_bf16_kernel<<<(NQ * 32) / 16, 256, 0, stream>>>(query,   Wq, bq, qb);
  proj_bf16_kernel<<<(NS * 32) / 16, 256, 0, stream>>>(support, Wk, bk, kb);

  // 2) Squared norms of the flattened (D*H) rows.
  rownorm_kernel<<<NQ, 256, 0, stream>>>(qb, q2);
  rownorm_kernel<<<NS, 256, 0, stream>>>(kb, k2);

  // 3) Main GEMM + distance epilogue (async double-buffered B panel).
  sim_kernel<<<dim3(NQ / 128, NS / 64), 256, 0, stream>>>(qb, kb, q2, k2, sim);

  // 4) Top-16 + softmax per query row (64KB dynamic LDS for the row).
  topk_kernel<<<NQ, 256, NS * sizeof(float), stream>>>(sim, idx_out, w_out);
}
